// MACEDiffusionAdapted_84894323572748
// MI455X (gfx1250) — matrix operand order repaired and checked by reading, not compiled, not run on hardware
//
#include <hip/hip_runtime.h>
#include <hip/hip_bf16.h>
#include <math.h>

typedef __attribute__((ext_vector_type(2)))  float    v2f;
typedef __attribute__((ext_vector_type(8)))  float    v8f;
typedef __attribute__((ext_vector_type(8)))  _Float16 v8h;
typedef __attribute__((ext_vector_type(16))) _Float16 v16h;

#define N_NODES 4096
#define E_EDGES 131072
#define C_CH    64
#define T_DIM   32
#define S_DIM   5
#define HR_DIM  64
#define SH_DIM  16
#define AVG_F   32.0f
#define EPS_F   1e-9f

// ---- workspace layout (float offsets) ----
static constexpr size_t OFF_POS    = 0;                                    // N*3
static constexpr size_t OFF_NF     = OFF_POS    + (size_t)N_NODES * 3;     // [n][c][x]  N*64*16
static constexpr size_t OFF_AGG    = OFF_NF     + (size_t)N_NODES * 1024;  // [x][n][c]  16*N*64
static constexpr size_t OFF_Y      = OFF_AGG    + (size_t)N_NODES * 1024;  // E*16
static constexpr size_t OFF_LEN0   = OFF_Y      + (size_t)E_EDGES * 16;    // E
static constexpr size_t OFF_R      = OFF_LEN0   + (size_t)E_EDGES;         // E*64 f32
static constexpr size_t OFF_H16    = OFF_R      + (size_t)E_EDGES * 64;    // E*64 halves
static constexpr size_t OFF_W2P    = OFF_H16    + (size_t)E_EDGES * 32;    // packed f16 B: 2*2*4*32*16 halves
static constexpr size_t OFF_WMIXP  = OFF_W2P    + 4096;                    // packed v2f B: 2*4*4*16*32*2
static constexpr size_t OFF_WPRODP = OFF_WMIXP  + 32768;                   // packed v2f B: 2*4*16*32*2

// ---------------------------------------------------------------------------
// prep: node embedding -> node_feats[:,:,0], zero rest, copy positions
// ---------------------------------------------------------------------------
__global__ void prep_kernel(const float* __restrict__ pos,
                            const int*   __restrict__ attrs,
                            const float* __restrict__ temb,
                            const float* __restrict__ embW,
                            const float* __restrict__ embB,
                            float* __restrict__ ws) {
    int n = blockIdx.x, c = threadIdx.x;
    float acc = embB[c];
    int a = attrs[n] - 1;
    acc += embW[(size_t)a * C_CH + c];
    #pragma unroll 4
    for (int t = 0; t < T_DIM; ++t)
        acc += temb[(size_t)n * T_DIM + t] * embW[(size_t)(S_DIM + t) * C_CH + c];
    float4* nf4 = (float4*)(ws + OFF_NF + ((size_t)n * C_CH + c) * SH_DIM);
    nf4[0] = make_float4(acc, 0.f, 0.f, 0.f);
    nf4[1] = make_float4(0.f, 0.f, 0.f, 0.f);
    nf4[2] = make_float4(0.f, 0.f, 0.f, 0.f);
    nf4[3] = make_float4(0.f, 0.f, 0.f, 0.f);
    if (c < 3) ws[OFF_POS + (size_t)n * 3 + c] = pos[(size_t)n * 3 + c];
}

// pack radial_W2 (f32 [L][64][64]) into f16 WMMA-B lane layout:
// dst[((l*2+kk)*4+ct)*32 + lane][i] = W2[l][kk*32 + (i<8 ? (lane>>4)*8 + i : 16 + (lane>>4)*8 + i-8)][ct*16 + (lane&15)]
__global__ void pack_w2_kernel(const float* __restrict__ w2, float* __restrict__ ws) {
    int idx = blockIdx.x * blockDim.x + threadIdx.x;   // 512 = 2*2*4*32
    if (idx >= 512) return;
    int lane = idx & 31, ct = (idx >> 5) & 3, kk = (idx >> 7) & 1, l = (idx >> 8) & 1;
    int g = lane >> 4, m = lane & 15;
    _Float16* dst = ((_Float16*)(ws + OFF_W2P)) + (size_t)idx * 16;
    #pragma unroll
    for (int i = 0; i < 16; ++i) {
        int K = kk * 32 + (i < 8 ? g * 8 + i : 16 + g * 8 + (i - 8)), n = ct * 16 + m;
        dst[i] = (_Float16)w2[((size_t)l * HR_DIM + K) * C_CH + n];
    }
}

// pack Wmix [L][4][64][64] into f32 WMMA-B (16x16x4) lane layout:
// dst[(((l*4+blk)*4+ct)*16+ks)*32 + lane][0..1] = W[ks*4+(lane>>4)*2 +{0,1}][ct*16+(lane&15)]
__global__ void pack_wmix_kernel(const float* __restrict__ wmix, float* __restrict__ ws) {
    int idx = blockIdx.x * blockDim.x + threadIdx.x;   // 16384 = 2*4*4*16*32
    if (idx >= 16384) return;
    int lane = idx & 31, ks = (idx >> 5) & 15, ct = (idx >> 9) & 3;
    int blk = (idx >> 11) & 3, l = (idx >> 13) & 1;
    int g = lane >> 4, m = lane & 15;
    int k = ks * 4 + g * 2, n = ct * 16 + m;
    const float* W = wmix + (size_t)(l * 4 + blk) * C_CH * C_CH;
    float* dst = ws + OFF_WMIXP + (size_t)idx * 2;
    dst[0] = W[(size_t)k * C_CH + n];
    dst[1] = W[(size_t)(k + 1) * C_CH + n];
}

// pack Wprod [L][64][64] similarly: idx over (l,ct,ks,lane) = 4096
__global__ void pack_wprod_kernel(const float* __restrict__ wprod, float* __restrict__ ws) {
    int idx = blockIdx.x * blockDim.x + threadIdx.x;
    if (idx >= 4096) return;
    int lane = idx & 31, ks = (idx >> 5) & 15, ct = (idx >> 9) & 3, l = (idx >> 11) & 1;
    int g = lane >> 4, m = lane & 15;
    int k = ks * 4 + g * 2, n = ct * 16 + m;
    const float* W = wprod + (size_t)l * C_CH * C_CH;
    float* dst = ws + OFF_WPRODP + (size_t)idx * 2;
    dst[0] = W[(size_t)k * C_CH + n];
    dst[1] = W[(size_t)(k + 1) * C_CH + n];
}

__global__ void zero_agg_kernel(float* __restrict__ ws) {
    size_t total = (size_t)N_NODES * 1024;
    for (size_t i = (size_t)blockIdx.x * blockDim.x + threadIdx.x; i < total;
         i += (size_t)gridDim.x * blockDim.x)
        ws[OFF_AGG + i] = 0.0f;
}

// ---------------------------------------------------------------------------
// per-edge geometry + spherical harmonics + radial hidden layer (silu, f16)
// ---------------------------------------------------------------------------
__global__ void edge_kernel(const int* __restrict__ eidx,
                            const float* __restrict__ W1,
                            const float* __restrict__ b1,
                            float* __restrict__ ws, int l, int store_len0) {
    int e = blockIdx.x, t = threadIdx.x;
    int s = eidx[e], r = eidx[E_EDGES + e];
    const float* P = ws + OFF_POS;
    float vx = P[(size_t)r * 3 + 0] - P[(size_t)s * 3 + 0];
    float vy = P[(size_t)r * 3 + 1] - P[(size_t)s * 3 + 1];
    float vz = P[(size_t)r * 3 + 2] - P[(size_t)s * 3 + 2];
    float len = sqrtf(vx * vx + vy * vy + vz * vz);
    float len0;
    if (store_len0) {
        len0 = len;
        if (t == 0) ws[OFF_LEN0 + e] = len;
    } else {
        len0 = ws[OFF_LEN0 + e];
    }
    float inv = 1.0f / (len + EPS_F);
    float x = vx * inv, y = vy * inv, z = vz * inv;

    if (t == 0) {
        const float s3  = 1.7320508075688772f;
        const float c2  = 3.872983346207417f;
        const float c20 = 1.118033988749895f;
        const float c22 = 1.9364916731037085f;
        const float c3a = 2.091650066335189f;
        const float c3b = 10.246950765959598f;
        const float c3c = 1.6201851746019651f;
        const float c30 = 1.3228756555322954f;
        const float c32 = 5.123475382979799f;
        float zz = z * z;
        float4* Y4 = (float4*)(ws + OFF_Y + (size_t)e * 16);
        Y4[0] = make_float4(1.0f, s3 * x, s3 * y, s3 * z);
        Y4[1] = make_float4(c2 * x * y, c2 * y * z, c20 * (3.0f * zz - 1.0f), c2 * x * z);
        Y4[2] = make_float4(c22 * (x * x - y * y),
                            c3a * y * (3.0f * x * x - y * y),
                            c3b * x * y * z,
                            c3c * y * (5.0f * zz - 1.0f));
        Y4[3] = make_float4(c30 * z * (5.0f * zz - 3.0f),
                            c3c * x * (5.0f * zz - 1.0f),
                            c32 * z * (x * x - y * y),
                            c3a * x * (x * x - 3.0f * y * y));
    }

    float zpre = len0 * W1[(size_t)(l * 2 + 0) * HR_DIM + t]
               + len  * W1[(size_t)(l * 2 + 1) * HR_DIM + t]
               + b1[(size_t)l * HR_DIM + t];
    float h = zpre / (1.0f + __expf(-zpre));
    ((_Float16*)(ws + OFF_H16))[(size_t)e * HR_DIM + t] = (_Float16)h;
}

// ---------------------------------------------------------------------------
// R = hidden(f16) @ W2(f16) -> f32 via v_wmma_f32_16x16x32_f16.
// A: two contiguous 16B loads per lane; B: pre-packed lane layout, two 16B loads.
// ---------------------------------------------------------------------------
__global__ void radial_gemm_kernel(float* __restrict__ ws, int l) {
    int lane = threadIdx.x;
    int g = lane >> 4, m = lane & 15;
    int e0 = blockIdx.x * 16;
    const _Float16* H  = (const _Float16*)(ws + OFF_H16);
    const _Float16* Wp = (const _Float16*)(ws + OFF_W2P);

    v8f acc[4] = {};
    #pragma unroll
    for (int kk = 0; kk < 2; ++kk) {
        const _Float16* hrow = H + (size_t)(e0 + m) * HR_DIM + kk * 32 + g * 8;
        v8h alo = *(const v8h*)hrow;
        v8h ahi = *(const v8h*)(hrow + 16);
        v16h a = __builtin_shufflevector(alo, ahi,
                                         0, 1, 2, 3, 4, 5, 6, 7,
                                         8, 9, 10, 11, 12, 13, 14, 15);
        #pragma unroll
        for (int ct = 0; ct < 4; ++ct) {
            const _Float16* bp = Wp + (size_t)(((l * 2 + kk) * 4 + ct) * 32 + lane) * 16;
            v8h blo = *(const v8h*)bp;
            v8h bhi = *(const v8h*)(bp + 8);
            v16h b = __builtin_shufflevector(blo, bhi,
                                             0, 1, 2, 3, 4, 5, 6, 7,
                                             8, 9, 10, 11, 12, 13, 14, 15);
            acc[ct] = __builtin_amdgcn_wmma_f32_16x16x32_f16(
                false, a, false, b, (short)0, acc[ct], false, false);
        }
    }
    float* R = ws + OFF_R;
    #pragma unroll
    for (int ct = 0; ct < 4; ++ct)
        #pragma unroll
        for (int j = 0; j < 8; ++j) {
            int row = j + g * 8;
            R[(size_t)(e0 + row) * C_CH + ct * 16 + m] = acc[ct][j];
        }
}

// ---------------------------------------------------------------------------
// scatter: m + tp into agg[x][recv][c] with f32 atomics
// ---------------------------------------------------------------------------
__global__ void scatter_kernel(const int* __restrict__ eidx, float* __restrict__ ws) {
    __shared__ float sY[16];
    int e = blockIdx.x, c = threadIdx.x;
    if (c < 4)
        ((float4*)sY)[c] = ((const float4*)(ws + OFF_Y + (size_t)e * 16))[c];
    __syncthreads();
    int snd = eidx[e], rcv = eidx[E_EDGES + e];
    const float4* h4 = (const float4*)(ws + OFF_NF + ((size_t)snd * C_CH + c) * SH_DIM);
    float4 ha = h4[0], hb = h4[1], hc = h4[2], hd = h4[3];
    float hv[16] = {ha.x, ha.y, ha.z, ha.w, hb.x, hb.y, hb.z, hb.w,
                    hc.x, hc.y, hc.z, hc.w, hd.x, hd.y, hd.z, hd.w};
    float r = ws[OFF_R + (size_t)e * C_CH + c];
    float dot = 0.0f;
    #pragma unroll
    for (int x = 0; x < 16; ++x) dot += hv[x] * sY[x];
    float* aggb = ws + OFF_AGG + (size_t)rcv * C_CH + c;   // [x][n][c] planes
    float scale = r * hv[0] * (1.0f / AVG_F);
    atomicAdd(aggb, (r * hv[0] * sY[0] + r * dot) * (1.0f / AVG_F));
    #pragma unroll
    for (int x = 1; x < 16; ++x)
        atomicAdd(aggb + (size_t)x * N_NODES * C_CH, scale * sY[x]);
}

// ---------------------------------------------------------------------------
// f32 16x16x4 WMMA wrapper
// ---------------------------------------------------------------------------
__device__ __forceinline__ v8f mma_f32_k4(v8f acc, v2f a, v2f b) {
    return __builtin_amdgcn_wmma_f32_16x16x4_f32(
        false, a, false, b, (short)0, acc, false, false);
}

// ---------------------------------------------------------------------------
// node mix: Wmix GEMMs -> gate -> Wprod GEMM -> Wvec -> pos update.
// One wave per 16-node tile; mixed tile in dynamic LDS laid out [x][node][ch].
// ---------------------------------------------------------------------------
__global__ void node_mix_kernel(const float* __restrict__ Wvec,
                                const float* __restrict__ pos_in,
                                float* __restrict__ ws,
                                float* __restrict__ out, int l, int last) {
    extern __shared__ float smem[];
    float* s_mixed = smem;                   // [16][16][64] = [x][node][ch]
    float* s_gate  = smem + 16 * 16 * 64;    // [16][64]
    float* s_nf3   = s_gate + 16 * 64;       // [16][64][3]
    int lane = threadIdx.x;
    int g = lane >> 4, m = lane & 15;
    int n0 = blockIdx.x * 16;
    const float* agg = ws + OFF_AGG;
    float* nf = ws + OFF_NF;

    // stage 1: mixed[x] = agg[x] @ Wmix[l, blk(x)]
    for (int x = 0; x < SH_DIM; ++x) {
        int blk = (x == 0) ? 0 : (x < 4) ? 1 : (x < 9) ? 2 : 3;
        const float* Bp = ws + OFF_WMIXP + (size_t)((l * 4 + blk) * 4) * 16 * 64;
        const float* Arow = agg + ((size_t)x * N_NODES + n0 + m) * C_CH + g * 2;
        for (int ct = 0; ct < 4; ++ct) {
            v8f acc = {};
            const float* bq = Bp + (size_t)ct * 16 * 64 + lane * 2;
            #pragma unroll
            for (int ks = 0; ks < 16; ++ks) {
                v2f a = *(const v2f*)(Arow + ks * 4);
                v2f b = *(const v2f*)(bq + ks * 64);
                acc = mma_f32_k4(acc, a, b);
            }
            #pragma unroll
            for (int j = 0; j < 8; ++j) {
                int M = j + g * 8;
                s_mixed[((size_t)x * 16 + M) * C_CH + ct * 16 + m] = acc[j];
            }
        }
    }
    __syncthreads();

    // gate(n,c) = s + s^2 + s^3 with s = mixed[x=0] (first 1024 floats)
    for (int i = lane; i < 16 * C_CH; i += 32) {
        float s = s_mixed[i];
        s_gate[i] = s + s * s + s * s * s;
    }
    __syncthreads();

    // stage 2: node_feats = mixed + (mixed*gate) @ Wprod[l]
    for (int x = 0; x < SH_DIM; ++x) {
        const float* mrow = s_mixed + ((size_t)x * 16 + m) * C_CH + g * 2;
        const float* grow = s_gate + (size_t)m * C_CH + g * 2;
        for (int ct = 0; ct < 4; ++ct) {
            v8f acc = {};
            const float* bq = ws + OFF_WPRODP + ((size_t)(l * 4 + ct) * 16) * 64 + lane * 2;
            #pragma unroll
            for (int ks = 0; ks < 16; ++ks) {
                v2f mm2 = *(const v2f*)(mrow + ks * 4);
                v2f gg2 = *(const v2f*)(grow + ks * 4);
                v2f a; a[0] = mm2[0] * gg2[0]; a[1] = mm2[1] * gg2[1];
                v2f b = *(const v2f*)(bq + ks * 64);
                acc = mma_f32_k4(acc, a, b);
            }
            #pragma unroll
            for (int j = 0; j < 8; ++j) {
                int M = j + g * 8;
                int d = ct * 16 + m;
                float v = s_mixed[((size_t)x * 16 + M) * C_CH + d] + acc[j];
                nf[((size_t)(n0 + M) * C_CH + d) * SH_DIM + x] = v;
                if (x >= 1 && x <= 3)
                    s_nf3[((size_t)M * C_CH + d) * 3 + (x - 1)] = v;
            }
        }
    }
    __syncthreads();

    // mbv = einsum('ncx,c->nx', nf[:,:,1:4], Wvec[l]); pos += mbv
    for (int o = lane; o < 48; o += 32) {
        int node = o / 3, j = o % 3;
        float acc = 0.0f;
        for (int c = 0; c < C_CH; ++c)
            acc += s_nf3[((size_t)node * C_CH + c) * 3 + j] * Wvec[(size_t)l * C_CH + c];
        size_t pidx = OFF_POS + (size_t)(n0 + node) * 3 + j;
        float p = ws[pidx] + acc;
        ws[pidx] = p;
        if (last)
            out[(size_t)(n0 + node) * 3 + j] = p - pos_in[(size_t)(n0 + node) * 3 + j];
    }
}

// ---------------------------------------------------------------------------
extern "C" void kernel_launch(void* const* d_in, const int* in_sizes, int n_in,
                              void* d_out, int out_size, void* d_ws, size_t ws_size,
                              hipStream_t stream) {
    const float* pos   = (const float*)d_in[0];
    const int*   attrs = (const int*)d_in[1];
    const float* temb  = (const float*)d_in[2];
    const int*   eidx  = (const int*)d_in[3];
    const float* embW  = (const float*)d_in[4];
    const float* embB  = (const float*)d_in[5];
    const float* W1    = (const float*)d_in[6];
    const float* b1    = (const float*)d_in[7];
    const float* W2    = (const float*)d_in[8];
    const float* Wmix  = (const float*)d_in[9];
    const float* Wprod = (const float*)d_in[10];
    const float* Wvec  = (const float*)d_in[11];
    float* out = (float*)d_out;
    float* ws  = (float*)d_ws;
    (void)in_sizes; (void)n_in; (void)out_size; (void)ws_size;

    prep_kernel<<<N_NODES, C_CH, 0, stream>>>(pos, attrs, temb, embW, embB, ws);
    pack_w2_kernel<<<2, 256, 0, stream>>>(W2, ws);
    pack_wmix_kernel<<<64, 256, 0, stream>>>(Wmix, ws);
    pack_wprod_kernel<<<16, 256, 0, stream>>>(Wprod, ws);

    const size_t lds_bytes = (16 * 16 * 64 + 16 * 64 + 16 * 64 * 3) * sizeof(float);
    for (int l = 0; l < 2; ++l) {
        zero_agg_kernel<<<2048, 256, 0, stream>>>(ws);
        edge_kernel<<<E_EDGES, HR_DIM, 0, stream>>>(eidx, W1, b1, ws, l, l == 0);
        radial_gemm_kernel<<<E_EDGES / 16, 32, 0, stream>>>(ws, l);
        scatter_kernel<<<E_EDGES, C_CH, 0, stream>>>(eidx, ws);
        node_mix_kernel<<<N_NODES / 16, 32, lds_bytes, stream>>>(
            Wvec, pos, ws, out, l, l == 1);
    }
}
